// GraphRecommender_89197880803439
// MI455X (gfx1250) — matrix operand order repaired and checked by reading, not compiled, not run on hardware
//
#include <hip/hip_runtime.h>
#include <hip/hip_bf16.h>
#include <math.h>

#define NNODES 100001
#define DIM    128
#define BATCH  512
#define SEQL   50
#define NEDGES 1200000
#define WK     12.0f
#define EPSN   1e-12f

typedef float    v8f  __attribute__((ext_vector_type(8)));
typedef _Float16 v16h __attribute__((ext_vector_type(16)));
typedef _Float16 v8h  __attribute__((ext_vector_type(8)));
typedef _Float16 v4h  __attribute__((ext_vector_type(4)));

// ---------------- zero-fill ----------------
__global__ void k_zero(float* __restrict__ p, int n) {
  int i = blockIdx.x * blockDim.x + threadIdx.x;
  if (i < n) p[i] = 0.f;
}

// ---------------- SpMM layer: dst[row] += val * src[col] ----------------
// one wave per edge; each lane owns 4 of the 128 dims (float4 gather + 4 atomics)
__global__ void k_spmm(const float* __restrict__ src, float* __restrict__ dst,
                       const int* __restrict__ row, const int* __restrict__ col,
                       const float* __restrict__ val) {
  int e    = blockIdx.x * (blockDim.x >> 5) + (threadIdx.x >> 5);
  int lane = threadIdx.x & 31;
  if (e >= NEDGES) return;
  int   r = row[e];
  int   c = col[e];
  float v = val[e];
  const float4* s4 = (const float4*)(src + (long)c * DIM);
  float4 x = s4[lane];
  float* d = dst + (long)r * DIM + lane * 4;
  atomicAdd(d + 0, v * x.x);
  atomicAdd(d + 1, v * x.y);
  atomicAdd(d + 2, v * x.z);
  atomicAdd(d + 3, v * x.w);
}

// ------- graph avg + l2 normalize -> f16 operand table (wave per node) ------
__global__ void k_gnorm(const float* __restrict__ emb, const float* __restrict__ h1,
                        const float* __restrict__ h2, _Float16* __restrict__ gn) {
  int node = blockIdx.x * (blockDim.x >> 5) + (threadIdx.x >> 5);
  int lane = threadIdx.x & 31;
  if (node >= NNODES) return;
  long base = (long)node * DIM + lane * 4;
  float4 e = *(const float4*)(emb + base);
  float4 a = *(const float4*)(h1 + base);
  float4 b = *(const float4*)(h2 + base);
  const float third = 1.f / 3.f;
  float4 g;
  g.x = (e.x + a.x + b.x) * third;
  g.y = (e.y + a.y + b.y) * third;
  g.z = (e.z + a.z + b.z) * third;
  g.w = (e.w + a.w + b.w) * third;
  float ss = g.x * g.x + g.y * g.y + g.z * g.z + g.w * g.w;
  #pragma unroll
  for (int m = 16; m >= 1; m >>= 1) ss += __shfl_xor(ss, m, 32);
  float inv = 1.f / fmaxf(sqrtf(ss), EPSN);
  v4h o = {(_Float16)(g.x * inv), (_Float16)(g.y * inv),
           (_Float16)(g.z * inv), (_Float16)(g.w * inv)};
  *(v4h*)(gn + base) = o;
}

// ---------------- seq_hidden gather + masked-mean hs (block per session) ---
__global__ void k_seqhs(const int* __restrict__ items, const int* __restrict__ inputs,
                        const int* __restrict__ alias, const float* __restrict__ emb,
                        const float* __restrict__ h1, const float* __restrict__ h2,
                        const float* __restrict__ feat, float* __restrict__ seq,
                        float* __restrict__ hs) {
  int b = blockIdx.x;
  int d = threadIdx.x;  // 0..127
  const float third = 1.f / 3.f;
  float acc = 0.f, cnt = 0.f;
  for (int l = 0; l < SEQL; ++l) {
    int  al = alias[b * SEQL + l];
    int  it = items[b * SEQL + al];
    long of = (long)it * DIM + d;
    float v = (emb[of] + h1[of] + h2[of]) * third + feat[of];
    seq[((long)(b * SEQL + l)) * DIM + d] = v;
    float m = (inputs[b * SEQL + l] != 0) ? 1.f : 0.f;
    acc += v * m;
    cnt += m;
  }
  hs[b * DIM + d] = acc / cnt;
}

// ---------------- hs_glu = hs @ glu2_w^T (block per session) ----------------
__global__ void k_hsglu(const float* __restrict__ hs, const float* __restrict__ glu2_w,
                        float* __restrict__ hsglu) {
  __shared__ float s_hs[DIM];
  int b = blockIdx.x;
  int d = threadIdx.x;
  s_hs[d] = hs[b * DIM + d];
  __syncthreads();
  float acc = 0.f;
  #pragma unroll 8
  for (int k = 0; k < DIM; ++k) acc += s_hs[k] * glu2_w[d * DIM + k];
  hsglu[b * DIM + d] = acc;
}

// ---------------- fused attention (block per (b,l)) ----------------
__global__ void k_attn(const int* __restrict__ inputs, const float* __restrict__ emb,
                       const float* __restrict__ seq, const float* __restrict__ hsglu,
                       const float* __restrict__ w_1, const float* __restrict__ w_2,
                       const float* __restrict__ glu1_w, const float* __restrict__ glu1_b,
                       float* __restrict__ sess) {
  __shared__ float s_pos[DIM];
  __shared__ float s_seq[DIM];
  __shared__ float s_nh[DIM];
  __shared__ float s_red[DIM];
  int bl = blockIdx.x;
  int b  = bl / SEQL;
  int l  = bl % SEQL;
  int d  = threadIdx.x;
  s_pos[d] = emb[(long)(SEQL - 1 - l) * DIM + d];  // reversed positional emb
  s_seq[d] = seq[(long)bl * DIM + d];
  __syncthreads();
  // nh1 = tanh([pos;seq] @ w_1)
  float acc = 0.f;
  #pragma unroll 8
  for (int k = 0; k < DIM; ++k) acc += s_pos[k] * w_1[k * DIM + d];
  #pragma unroll 8
  for (int k = 0; k < DIM; ++k) acc += s_seq[k] * w_1[(DIM + k) * DIM + d];
  float nh1 = tanhf(acc);
  s_nh[d] = nh1;
  __syncthreads();
  // nh2 = sigmoid(nh1 @ glu1_w^T + glu1_b + hs_glu)
  float acc2 = glu1_b[d] + hsglu[b * DIM + d];
  #pragma unroll 8
  for (int k = 0; k < DIM; ++k) acc2 += s_nh[k] * glu1_w[d * DIM + k];
  float nh2 = 1.f / (1.f + expf(-acc2));
  // beta = dot(nh2, w_2) * mask
  s_red[d] = nh2 * w_2[d];
  __syncthreads();
  #pragma unroll
  for (int s = 64; s > 0; s >>= 1) {
    if (d < s) s_red[d] += s_red[d + s];
    __syncthreads();
  }
  float mask = (inputs[b * SEQL + l] != 0) ? 1.f : 0.f;
  float beta = s_red[0] * mask;
  atomicAdd(&sess[b * DIM + d], beta * s_seq[d]);
}

// -------- select = W_K * l2norm(sess) -> f16 operand table (wave/session) --
__global__ void k_select(const float* __restrict__ sess, _Float16* __restrict__ sel) {
  int b    = blockIdx.x * (blockDim.x >> 5) + (threadIdx.x >> 5);
  int lane = threadIdx.x & 31;
  if (b >= BATCH) return;
  long base = (long)b * DIM + lane * 4;
  float4 g = *(const float4*)(sess + base);
  float ss = g.x * g.x + g.y * g.y + g.z * g.z + g.w * g.w;
  #pragma unroll
  for (int m = 16; m >= 1; m >>= 1) ss += __shfl_xor(ss, m, 32);
  float inv = WK / fmaxf(sqrtf(ss), EPSN);
  v4h o = {(_Float16)(g.x * inv), (_Float16)(g.y * inv),
           (_Float16)(g.z * inv), (_Float16)(g.w * inv)};
  *(v4h*)(sel + base) = o;
}

// ------------- WMMA scoring GEMM: scores = select @ gnorm^T ----------------
// One wave computes a 16x64 strip (4 N-tiles). Per K-step of 32: clause-load
// the A fragment + all four B fragments into distinct register sets, then
// issue 4 back-to-back v_wmma_f32_16x16x32_f16 (f32 accumulate).
__global__ void k_scores(const _Float16* __restrict__ sel,
                         const _Float16* __restrict__ gn,
                         float* __restrict__ out) {
  const int NSTRIP = (NNODES + 63) / 64;  // 1563 strips of 64 columns
  int wave  = threadIdx.x >> 5;
  int lane  = threadIdx.x & 31;
  int strip = blockIdx.x * (blockDim.x >> 5) + wave;
  if (strip >= NSTRIP) return;            // uniform per-wave exit; EXEC stays full
  int m0   = blockIdx.y * 16;
  int n0   = strip * 64;
  int half = lane >> 4;
  int lr   = lane & 15;
  // A fragment (16x32 f16): lane holds A[m0+lr][k0+8h..+7] and [k0+16+8h..+7]
  const _Float16* Arow = sel + (long)(m0 + lr) * DIM;
  // B fragment (32x16 f16, B = gn^T): lane holds gn[n][k0+8h..+7] and [k0+16+8h..+7]
  int nidx[4];
  const _Float16* Brow[4];
  #pragma unroll
  for (int t = 0; t < 4; ++t) {
    int n = n0 + 16 * t + lr;
    nidx[t] = n;
    Brow[t] = gn + (long)((n < NNODES) ? n : 0) * DIM;
  }
  v8f c[4];
  #pragma unroll
  for (int t = 0; t < 4; ++t) c[t] = (v8f){0.f, 0.f, 0.f, 0.f, 0.f, 0.f, 0.f, 0.f};

  #pragma unroll
  for (int k0 = 0; k0 < DIM; k0 += 32) {
    v8h alo = *(const v8h*)(Arow + k0 + 8 * half);
    v8h ahi = *(const v8h*)(Arow + k0 + 16 + 8 * half);
    v16h a;
    #pragma unroll
    for (int i = 0; i < 8; ++i) { a[i] = alo[i]; a[i + 8] = ahi[i]; }
    // materialize all four B fragments first -> distinct dest registers,
    // lets the compiler clause the loads and throttle with partial waits
    v16h bf[4];
    #pragma unroll
    for (int t = 0; t < 4; ++t) {
      v8h blo = *(const v8h*)(Brow[t] + k0 + 8 * half);
      v8h bhi = *(const v8h*)(Brow[t] + k0 + 16 + 8 * half);
      #pragma unroll
      for (int i = 0; i < 8; ++i) { bf[t][i] = blo[i]; bf[t][i + 8] = bhi[i]; }
    }
    #pragma unroll
    for (int t = 0; t < 4; ++t) {
      c[t] = __builtin_amdgcn_wmma_f32_16x16x32_f16(
          false, a, false, bf[t], (short)0, c[t], false, false);
    }
  }
  // C/D layout: lanes 0-15 -> M = m0+r, N = n; lanes 16-31 -> M = m0+8+r
  #pragma unroll
  for (int t = 0; t < 4; ++t) {
    if (nidx[t] < NNODES) {
      #pragma unroll
      for (int r = 0; r < 8; ++r) {
        int m = m0 + r + 8 * half;
        out[(long)m * NNODES + nidx[t]] = c[t][r];
      }
    }
  }
}

extern "C" void kernel_launch(void* const* d_in, const int* in_sizes, int n_in,
                              void* d_out, int out_size, void* d_ws, size_t ws_size,
                              hipStream_t stream) {
  const int*   items   = (const int*)d_in[0];
  const int*   inputs  = (const int*)d_in[1];
  const int*   alias   = (const int*)d_in[2];
  const float* feat    = (const float*)d_in[3];
  const int*   adj_row = (const int*)d_in[4];
  const int*   adj_col = (const int*)d_in[5];
  const float* adj_val = (const float*)d_in[6];
  const float* emb     = (const float*)d_in[7];
  const float* w_1     = (const float*)d_in[8];
  const float* w_2     = (const float*)d_in[9];
  const float* glu1_w  = (const float*)d_in[10];
  const float* glu1_b  = (const float*)d_in[11];
  const float* glu2_w  = (const float*)d_in[12];
  float* scores = (float*)d_out;

  // workspace layout (byte-carved; mixed f32 / f16 regions)
  char* p = (char*)d_ws;
  float* h1 = (float*)p;          p += (size_t)NNODES * DIM * sizeof(float);
  float* h2 = (float*)p;          p += (size_t)NNODES * DIM * sizeof(float);
  _Float16* gnh = (_Float16*)p;   p += (size_t)NNODES * DIM * sizeof(_Float16);
  float* seq = (float*)p;         p += (size_t)BATCH * SEQL * DIM * sizeof(float);
  float* hs = (float*)p;          p += (size_t)BATCH * DIM * sizeof(float);
  float* hsglu = (float*)p;       p += (size_t)BATCH * DIM * sizeof(float);
  float* sess = (float*)p;        p += (size_t)BATCH * DIM * sizeof(float);
  _Float16* selh = (_Float16*)p;  p += (size_t)BATCH * DIM * sizeof(_Float16);

  // zero accumulators (h1,h2 contiguous; sess separate)
  {
    int n = 2 * NNODES * DIM;
    k_zero<<<(n + 255) / 256, 256, 0, stream>>>(h1, n);
    int m = BATCH * DIM;
    k_zero<<<(m + 255) / 256, 256, 0, stream>>>(sess, m);
  }

  // 2-layer sparse propagation (L2-resident feature table)
  {
    dim3 grid((NEDGES + 7) / 8);
    k_spmm<<<grid, 256, 0, stream>>>(emb, h1, adj_row, adj_col, adj_val);
    k_spmm<<<grid, 256, 0, stream>>>(h1, h2, adj_row, adj_col, adj_val);
  }

  // average + l2-normalize node embeddings -> f16 operand table
  k_gnorm<<<(NNODES + 7) / 8, 256, 0, stream>>>(emb, h1, h2, gnh);

  // session feature gather + masked mean
  k_seqhs<<<BATCH, DIM, 0, stream>>>(items, inputs, alias, emb, h1, h2, feat, seq, hs);

  // hs @ glu2_w^T
  k_hsglu<<<BATCH, DIM, 0, stream>>>(hs, glu2_w, hsglu);

  // fused attention + session-embedding accumulation
  k_attn<<<BATCH * SEQL, DIM, 0, stream>>>(inputs, emb, seq, hsglu, w_1, w_2,
                                           glu1_w, glu1_b, sess);

  // select = W_K * l2norm(sess) -> f16 operand table
  k_select<<<(BATCH + 7) / 8, 256, 0, stream>>>(sess, selh);

  // WMMA scoring GEMM: [512 x 100001] in 16x64 strips per wave
  {
    const int NSTRIP = (NNODES + 63) / 64;  // 1563
    dim3 grid((NSTRIP + 7) / 8, BATCH / 16);
    k_scores<<<grid, 256, 0, stream>>>(selh, gnh, scores);
  }
}
